// Attention_44951127720570
// MI455X (gfx1250) — compile-verified, matrix operand order checked by reading, not run on hardware
//
#include <hip/hip_runtime.h>
#include <cstdint>

// ---------------- problem constants ----------------
#define B_      2
#define S_      1024
#define D_      4096
#define HQ_     32
#define HKV_    8
#define HD_     128
#define GROUPS_ (HQ_ / HKV_)
#define MTOT    (B_ * S_)            // 2048 rows for all big GEMMs
#define NQ_     (HQ_ * HD_)          // 4096
#define NKV_    (HKV_ * HD_)         // 1024
#define SCALE_  0.08838834764831845f // 1/sqrt(128)

typedef unsigned short u16;
typedef float  v8f   __attribute__((ext_vector_type(8)));
typedef __bf16 v16bf __attribute__((ext_vector_type(16)));
typedef int    v4i   __attribute__((vector_size(16)));   // matches builtin param type

struct alignas(16) U128 { unsigned int a, b, c, d; };
union FragBF { U128 u4[2]; v16bf v; };   // 32 bytes = one 16x32 bf16 fragment per lane

__device__ __forceinline__ u16 f2bf(float f) {
    unsigned int u = __float_as_uint(f);
    u += 0x7FFFu + ((u >> 16) & 1u);     // round-to-nearest-even
    return (u16)(u >> 16);
}

__device__ __forceinline__ v8f wmma_bf16(v16bf a, v16bf b, v8f c) {
    // D = A(16x32) * B(32x16) + C(16x16), f32 accum
    return __builtin_amdgcn_wmma_f32_16x16x32_bf16(false, a, false, b, (short)0, c,
                                                   false, false);
}

// ---------------- CDNA5 async global->LDS path (guarded) ----------------
#if defined(__AMDGCN__) &&                                            \
    __has_builtin(__builtin_amdgcn_global_load_async_to_lds_b128) &&  \
    __has_builtin(__builtin_amdgcn_s_wait_asynccnt)
#define HAVE_ASYNC_LDS 1
#else
#define HAVE_ASYNC_LDS 0
#endif

__device__ __forceinline__ void async_cp16(const u16* g, u16* l) {
#if HAVE_ASYNC_LDS
    __builtin_amdgcn_global_load_async_to_lds_b128(
        (__attribute__((address_space(1))) v4i*)g,
        (__attribute__((address_space(3))) v4i*)l, 0, 0);
#else
    *(U128*)l = *(const U128*)g;
#endif
}
__device__ __forceinline__ void async_wait_4() {
#if HAVE_ASYNC_LDS
    __builtin_amdgcn_s_wait_asynccnt(4);
#endif
}
__device__ __forceinline__ void async_wait_0() {
#if HAVE_ASYNC_LDS
    __builtin_amdgcn_s_wait_asynccnt(0);
#endif
}

// ---------------- kernel 1: f32 -> bf16 convert (row-major preserved) ----------------
__global__ void __launch_bounds__(256) k_convert_bf16(const float* __restrict__ in,
                                                      u16* __restrict__ out, int n) {
    int i = (blockIdx.x * 256 + threadIdx.x) * 4;
    if (i + 3 < n) {
        float4 v = *(const float4*)(in + i);
        out[i + 0] = f2bf(v.x);
        out[i + 1] = f2bf(v.y);
        out[i + 2] = f2bf(v.z);
        out[i + 3] = f2bf(v.w);
    }
}

// ---------------- kernel 2: f32 [R][C] -> bf16 transposed [C][R] ----------------
__global__ void __launch_bounds__(256) k_transpose_bf16(const float* __restrict__ in,
                                                        u16* __restrict__ out,
                                                        int R, int C) {
    __shared__ u16 tile[64][65];
    int r0 = blockIdx.y * 64, c0 = blockIdx.x * 64;
    int t = threadIdx.x;
    for (int i = t; i < 64 * 64; i += 256) {
        int r = i >> 6, c = i & 63;
        tile[r][c] = f2bf(in[(size_t)(r0 + r) * C + c0 + c]);
    }
    __syncthreads();
    for (int i = t; i < 64 * 64; i += 256) {
        int c = i >> 6, r = i & 63;  // write rows of the transposed matrix, coalesced in r
        out[(size_t)(c0 + c) * R + r0 + r] = tile[r][c];
    }
}

// ---------------- kernel 3: C[M][N] f32 = A[M][K] bf16  x  (Bt[N][K] bf16)^T ----------
// Double-buffered LDS staging via async global->LDS (ASYNCcnt) when available.
#define BM 128
#define BN 128
#define BKK 32
__global__ void __launch_bounds__(256) k_gemm_tn(const u16* __restrict__ A,
                                                 const u16* __restrict__ Bt,
                                                 float* __restrict__ C,
                                                 int M, int N, int K) {
    __shared__ u16 As[2][BM][BKK];   // 2 x 8 KB
    __shared__ u16 Bs[2][BN][BKK];   // 2 x 8 KB
    int t = threadIdx.x, lane = t & 31, w = t >> 5;
    int m0 = blockIdx.y * BM, n0 = blockIdx.x * BN;
    int half = (lane >> 4) * 8;        // fragment k-octet select per lane half
    int l15  = lane & 15;
    int ar   = w * 16 + l15;           // this wave's A row within the 128-row tile

    // staging slice for this thread: rows r0 and r0+64, 16B column chunk c0
    int r0 = t >> 2, c0 = (t & 3) * 8;

    auto stage = [&](int buf, int kb) {
        async_cp16(&A [(size_t)(m0 + r0)      * K + kb + c0], &As[buf][r0]     [c0]);
        async_cp16(&A [(size_t)(m0 + r0 + 64) * K + kb + c0], &As[buf][r0 + 64][c0]);
        async_cp16(&Bt[(size_t)(n0 + r0)      * K + kb + c0], &Bs[buf][r0]     [c0]);
        async_cp16(&Bt[(size_t)(n0 + r0 + 64) * K + kb + c0], &Bs[buf][r0 + 64][c0]);
    };

    v8f acc[8] = {};
    auto compute = [&](int buf) {
        FragBF a;
        a.u4[0] = *(U128*)&As[buf][ar][half];
        a.u4[1] = *(U128*)&As[buf][ar][16 + half];
        for (int g = 0; g < 2; ++g) {
            FragBF b[4];
            for (int j = 0; j < 4; ++j) {           // batch 4 B-fragments, then 4 WMMAs
                int br = (g * 4 + j) * 16 + l15;
                b[j].u4[0] = *(U128*)&Bs[buf][br][half];
                b[j].u4[1] = *(U128*)&Bs[buf][br][16 + half];
            }
            for (int j = 0; j < 4; ++j)
                acc[g * 4 + j] = wmma_bf16(a.v, b[j].v, acc[g * 4 + j]);
        }
    };

    int nsteps = K / BKK;
    stage(0, 0);                               // prologue: 4 async b128 in flight
    for (int s = 0; s < nsteps - 1; ++s) {
        int cur = s & 1;
        stage(1 - cur, (s + 1) * BKK);         // stream next slab while computing
        async_wait_4();                        // in-order: current slab complete
        __syncthreads();
        compute(cur);
        __syncthreads();
    }
    {
        int cur = (nsteps - 1) & 1;
        async_wait_0();
        __syncthreads();
        compute(cur);
    }

    for (int tn = 0; tn < 8; ++tn) {
        int col = n0 + tn * 16 + l15;
        for (int e = 0; e < 8; ++e) {
            int row = m0 + w * 16 + e + ((lane >> 4) << 3);
            C[(size_t)row * N + col] = acc[tn][e];
        }
    }
}

// ---------------- kernel 4: RoPE + head-major transpose (f32 in, bf16 out) -----------
// in:  [b][s][h][d] f32  (GEMM output), out: [b][h][s][d] bf16
__global__ void __launch_bounds__(128) k_rope(const float* __restrict__ X,
                                              const float* __restrict__ cosT,
                                              const float* __restrict__ sinT,
                                              u16* __restrict__ Out, int H) {
    int d = threadIdx.x;
    int idx = blockIdx.x;                 // (b*S + s)*H + h
    int h = idx % H;
    int bs = idx / H;
    int s = bs % S_, b = bs / S_;
    const float* xp = X + (size_t)idx * HD_;
    float x = xp[d];
    float other = (d < 64) ? -xp[d + 64] : xp[d - 64];
    float c  = cosT[(size_t)bs * HD_ + d];
    float sn = sinT[(size_t)bs * HD_ + d];
    Out[(((size_t)b * H + h) * S_ + s) * HD_ + d] = f2bf(x * c + other * sn);
}

// ---------------- kernel 5: V reshape to [b][h][s][d] bf16 ----------------
__global__ void __launch_bounds__(128) k_vshape(const float* __restrict__ X,
                                                u16* __restrict__ Out) {
    int d = threadIdx.x;
    int idx = blockIdx.x;                 // (b*S + s)*HKV + h
    int h = idx % HKV_;
    int bs = idx / HKV_;
    int s = bs % S_, b = bs / S_;
    Out[(((size_t)b * HKV_ + h) * S_ + s) * HD_ + d] = f2bf(X[(size_t)idx * HD_ + d]);
}

// ---------------- kernel 6: flash attention (bf16 WMMA, f32 softmax) -----------------
#define BQ  32
#define BKV 64
__global__ void __launch_bounds__(256) k_flash(const u16* __restrict__ Qt,
                                               const u16* __restrict__ Kt,
                                               const u16* __restrict__ Vt,
                                               u16* __restrict__ AO) {
    __shared__ u16   Qs [BQ][HD_];    // 8 KB
    __shared__ u16   Ks [BKV][HD_];   // 16 KB
    __shared__ u16   Vts[HD_][BKV];   // 16 KB (transposed: [d][kv])
    __shared__ float Sc [BQ][BKV];    // 8 KB
    __shared__ u16   Ps [BQ][BKV];    // 4 KB
    __shared__ float mrow[BQ], lrow[BQ], arow[BQ];

    const int nqb = S_ / BQ;
    int qb = blockIdx.x % nqb;
    int bh = blockIdx.x / nqb;
    int h = bh % HQ_, b = bh / HQ_;
    int hkv = h / GROUPS_;

    const u16* Qbase = Qt + (((size_t)(b * HQ_  + h  ) * S_) + qb * BQ) * HD_;
    const u16* Kbase = Kt +  ((size_t)(b * HKV_ + hkv) * S_) * HD_;
    const u16* Vbase = Vt +  ((size_t)(b * HKV_ + hkv) * S_) * HD_;

    int t = threadIdx.x, lane = t & 31, w = t >> 5;
    int half = (lane >> 4) * 8, l15 = lane & 15;
    int mt = w & 1;          // q-tile (2 tiles of 16 over BQ=32)
    int wd = w >> 1;         // d-tile group (4 groups x 2 = 8 d-tiles of 16)

    for (int i = t; i < BQ * HD_ / 8; i += 256) {
        int r = i >> 4, c8 = (i & 15) * 8;
        *(U128*)&Qs[r][c8] = *(const U128*)&Qbase[(size_t)r * HD_ + c8];
    }
    if (t < BQ) { mrow[t] = -1e30f; lrow[t] = 0.0f; }
    v8f o[2] = {};
    __syncthreads();

    for (int kc = 0; kc < S_; kc += BKV) {
        // ---- stage K rows (row-major) and V rows transposed into LDS ----
        for (int i = t; i < BKV * HD_ / 8; i += 256) {
            int r = i >> 4, c8 = (i & 15) * 8;
            *(U128*)&Ks[r][c8] = *(const U128*)&Kbase[(size_t)(kc + r) * HD_ + c8];
            U128 x = *(const U128*)&Vbase[(size_t)(kc + r) * HD_ + c8];
            const u16* e = (const u16*)&x;
            for (int j = 0; j < 8; ++j) Vts[c8 + j][r] = e[j];
        }
        __syncthreads();

        // ---- scores: wave w computes tile (mt, nt) of Q K^T ----
        {
            int nt = w >> 1;                 // 4 kv-tiles of 16
            int qrow = mt * 16 + l15;
            int krow = nt * 16 + l15;
            FragBF aa[4], bb[4];
            for (int k8 = 0; k8 < 4; ++k8) { // batch all fragment loads first
                int kk = k8 * 32;
                aa[k8].u4[0] = *(U128*)&Qs[qrow][kk + half];
                aa[k8].u4[1] = *(U128*)&Qs[qrow][kk + 16 + half];
                bb[k8].u4[0] = *(U128*)&Ks[krow][kk + half];
                bb[k8].u4[1] = *(U128*)&Ks[krow][kk + 16 + half];
            }
            v8f s = {};
            for (int k8 = 0; k8 < 4; ++k8) s = wmma_bf16(aa[k8].v, bb[k8].v, s);

            int scol = nt * 16 + l15;
            for (int e = 0; e < 8; ++e) {
                int srow = mt * 16 + e + ((lane >> 4) << 3);
                Sc[srow][scol] = s[e] * SCALE_;
            }
        }
        __syncthreads();

        // ---- online softmax per row ----
        if (t < BQ) {
            float mo = mrow[t], cm = mo;
            for (int j = 0; j < BKV; ++j) cm = fmaxf(cm, Sc[t][j]);
            float alpha = __expf(mo - cm);
            float sum = 0.0f;
            for (int j = 0; j < BKV; ++j) {
                float p = __expf(Sc[t][j] - cm);
                sum += p;
                Ps[t][j] = f2bf(p);
            }
            mrow[t] = cm;
            lrow[t] = lrow[t] * alpha + sum;
            arow[t] = alpha;
        }
        __syncthreads();

        // ---- O = O*alpha + P @ V_chunk ----
        for (int u = 0; u < 2; ++u) {
            int dt = wd * 2 + u;
            v8f acc = o[u];
            for (int e = 0; e < 8; ++e) {
                int srow = mt * 16 + e + ((lane >> 4) << 3);
                acc[e] *= arow[srow];
            }
            int prow = mt * 16 + l15;
            int drow = dt * 16 + l15;
            FragBF aa[2], bb[2];
            for (int k8 = 0; k8 < 2; ++k8) {
                int kk = k8 * 32;
                aa[k8].u4[0] = *(U128*)&Ps[prow][kk + half];
                aa[k8].u4[1] = *(U128*)&Ps[prow][kk + 16 + half];
                bb[k8].u4[0] = *(U128*)&Vts[drow][kk + half];
                bb[k8].u4[1] = *(U128*)&Vts[drow][kk + 16 + half];
            }
            for (int k8 = 0; k8 < 2; ++k8) acc = wmma_bf16(aa[k8].v, bb[k8].v, acc);
            o[u] = acc;
        }
        __syncthreads();
    }

    // ---- normalize and store bf16 attn output [b][s][h*HD+d] ----
    for (int u = 0; u < 2; ++u) {
        int dt = wd * 2 + u;
        int col = h * HD_ + dt * 16 + l15;
        for (int e = 0; e < 8; ++e) {
            int srow = mt * 16 + e + ((lane >> 4) << 3);
            int qg = qb * BQ + srow;
            float val = o[u][e] / lrow[srow];
            AO[(size_t)(b * S_ + qg) * NQ_ + col] = f2bf(val);
        }
    }
}

// ---------------- workspace layout ----------------
static constexpr size_t OFF_XB  = 0;                                  // 2048*4096 bf16
static constexpr size_t OFF_WQT = OFF_XB  + (size_t)MTOT * D_ * 2;    // 4096*4096 bf16
static constexpr size_t OFF_WKT = OFF_WQT + (size_t)NQ_  * D_ * 2;    // 1024*4096 bf16
static constexpr size_t OFF_WVT = OFF_WKT + (size_t)NKV_ * D_ * 2;
static constexpr size_t OFF_WOT = OFF_WVT + (size_t)NKV_ * D_ * 2;    // 4096*4096 bf16
static constexpr size_t OFF_QF  = OFF_WOT + (size_t)D_   * NQ_ * 2;   // 2048*4096 f32
static constexpr size_t OFF_KF  = OFF_QF  + (size_t)MTOT * NQ_ * 4;   // 2048*1024 f32
static constexpr size_t OFF_VF  = OFF_KF  + (size_t)MTOT * NKV_ * 4;
static constexpr size_t OFF_QT  = OFF_VF  + (size_t)MTOT * NKV_ * 4;  // bf16 [b][h][s][d]
static constexpr size_t OFF_KT  = OFF_QT  + (size_t)MTOT * NQ_ * 2;
static constexpr size_t OFF_VT  = OFF_KT  + (size_t)MTOT * NKV_ * 2;
static constexpr size_t OFF_AO  = OFF_QF;  // reuse Qf32 region after RoPE (16MB <= 32MB)

extern "C" void kernel_launch(void* const* d_in, const int* in_sizes, int n_in,
                              void* d_out, int out_size, void* d_ws, size_t ws_size,
                              hipStream_t stream) {
    const float* hs   = (const float*)d_in[0];
    const float* cosT = (const float*)d_in[1];
    const float* sinT = (const float*)d_in[2];
    const float* Wq   = (const float*)d_in[3];
    const float* Wk   = (const float*)d_in[4];
    const float* Wv   = (const float*)d_in[5];
    const float* Wo   = (const float*)d_in[6];
    float* out = (float*)d_out;

    char* ws = (char*)d_ws;
    u16*   Xb  = (u16*)  (ws + OFF_XB);
    u16*   WqT = (u16*)  (ws + OFF_WQT);
    u16*   WkT = (u16*)  (ws + OFF_WKT);
    u16*   WvT = (u16*)  (ws + OFF_WVT);
    u16*   WoT = (u16*)  (ws + OFF_WOT);
    float* Qf  = (float*)(ws + OFF_QF);
    float* Kf  = (float*)(ws + OFF_KF);
    float* Vf  = (float*)(ws + OFF_VF);
    u16*   Qt  = (u16*)  (ws + OFF_QT);
    u16*   Kt  = (u16*)  (ws + OFF_KT);
    u16*   Vt  = (u16*)  (ws + OFF_VT);
    u16*   AO  = (u16*)  (ws + OFF_AO);

    // 1) precision conversion / weight transposes
    {
        int n = MTOT * D_;
        k_convert_bf16<<<n / 1024, 256, 0, stream>>>(hs, Xb, n);
    }
    k_transpose_bf16<<<dim3(NQ_  / 64, D_ / 64), 256, 0, stream>>>(Wq, WqT, D_,  NQ_);
    k_transpose_bf16<<<dim3(NKV_ / 64, D_ / 64), 256, 0, stream>>>(Wk, WkT, D_,  NKV_);
    k_transpose_bf16<<<dim3(NKV_ / 64, D_ / 64), 256, 0, stream>>>(Wv, WvT, D_,  NKV_);
    k_transpose_bf16<<<dim3(D_   / 64, NQ_ / 64), 256, 0, stream>>>(Wo, WoT, NQ_, D_);

    // 2) Q/K/V projections (WMMA)
    k_gemm_tn<<<dim3(NQ_  / BN, MTOT / BM), 256, 0, stream>>>(Xb, WqT, Qf, MTOT, NQ_,  D_);
    k_gemm_tn<<<dim3(NKV_ / BN, MTOT / BM), 256, 0, stream>>>(Xb, WkT, Kf, MTOT, NKV_, D_);
    k_gemm_tn<<<dim3(NKV_ / BN, MTOT / BM), 256, 0, stream>>>(Xb, WvT, Vf, MTOT, NKV_, D_);

    // 3) RoPE + layout to [b][h][s][d] bf16
    k_rope  <<<MTOT * HQ_,  HD_, 0, stream>>>(Qf, cosT, sinT, Qt, HQ_);
    k_rope  <<<MTOT * HKV_, HD_, 0, stream>>>(Kf, cosT, sinT, Kt, HKV_);
    k_vshape<<<MTOT * HKV_, HD_, 0, stream>>>(Vf, Vt);

    // 4) flash attention (WMMA) -> bf16 [b][s][HQ*HD]
    k_flash<<<B_ * HQ_ * (S_ / BQ), 256, 0, stream>>>(Qt, Kt, Vt, AO);

    // 5) output projection (WMMA) -> f32 d_out
    k_gemm_tn<<<dim3(D_ / BN, MTOT / BM), 256, 0, stream>>>(AO, WoT, out, MTOT, D_, NQ_);

    (void)in_sizes; (void)n_in; (void)out_size; (void)ws_size;
}
